// DecoderModel_45449343926564
// MI455X (gfx1250) — matrix operand order repaired and checked by reading, not compiled, not run on hardware
//
#include <hip/hip_runtime.h>
#include <hip/hip_bf16.h>
#include <math.h>

// ---------------- model dims ----------------
#define DD    512
#define KK    128
#define VV    128
#define VOCAB 10001
#define BB    64
#define TDEC  192
#define TENC  512
#define QT    4      // N-tiles per wave in the WMMA GEMM

typedef __bf16 bf16_t;
typedef __bf16 v16bf __attribute__((ext_vector_type(16)));
typedef __bf16 v8bf  __attribute__((ext_vector_type(8)));
typedef float  v8f   __attribute__((ext_vector_type(8)));

// =====================================================================
// WMMA bf16 GEMM:  C(MxN) = act( A0(MxK0)*W0(NxK0)^T
//                              + A1(MxK1)*W1(NxK1)^T + bias )
// A* and W* are bf16 row-major.  One wave owns one 16-row strip and
// QT=4 adjacent 16-col tiles: each A fragment (2x 16B loads) feeds 4
// back-to-back V_WMMA_F32_16X16X32_BF16.  K0/K1 multiples of 32.
// Output: f32 (C!=null) and/or bf16 (Cbf!=null). mode1 = leaky_relu.
// =====================================================================
__device__ __forceinline__ v16bf load_a_frag(const bf16_t* __restrict__ ap,
                                             int k0, int lane) {
    const int kb = (lane < 16) ? 0 : 16;                // contiguous k = 16h + s, matches B
    v8bf lo = *(const v8bf*)(ap + k0 + kb);
    v8bf hi = *(const v8bf*)(ap + k0 + kb + 8);
    return __builtin_shufflevector(lo, hi, 0,1,2,3,4,5,6,7,8,9,10,11,12,13,14,15);
}

__device__ __forceinline__ void wmma_quad_k(v8f acc[QT],
    const bf16_t* __restrict__ A, int lda, int Kdim,
    const bf16_t* __restrict__ W, size_t ldw,
    int arow, const int wrow[QT], int lane)
{
    const int wk = (lane < 16) ? 0 : 16;                // contiguous k = 16h + s, same as A
    const bf16_t* ap = A + (size_t)arow * lda;
    const bf16_t* wp[QT];
#pragma unroll
    for (int j = 0; j < QT; ++j) wp[j] = W + (size_t)wrow[j] * ldw + wk;

    for (int k0 = 0; k0 < Kdim; k0 += 32) {
        v16bf a = load_a_frag(ap, k0, lane);
#pragma unroll
        for (int j = 0; j < QT; ++j) {
            __builtin_prefetch(wp[j] + k0 + 128, 0, 1); // next K-blocks from L2
            v16bf b = *(const v16bf*)(wp[j] + k0);
            acc[j] = __builtin_amdgcn_wmma_f32_16x16x32_bf16(
                         false, a, false, b, (short)0, acc[j], false, false);
        }
    }
}

__global__ void wmma_gemm_dual(
    const bf16_t* __restrict__ A0, int lda0, int K0,
    const bf16_t* __restrict__ W0, int ldw0,
    const bf16_t* __restrict__ A1, int lda1, int K1,
    const bf16_t* __restrict__ W1, int ldw1,
    const float* __restrict__ bias,
    float* __restrict__ C, bf16_t* __restrict__ Cbf, int ldc,
    int M, int N, int mode)
{
    const int lane   = threadIdx.x & 31;
    const int wave   = (blockIdx.x * (blockDim.x >> 5)) + (threadIdx.x >> 5);
    const int ntiles = (N + 15) >> 4;
    const int nquads = (ntiles + QT - 1) / QT;
    const int mtiles = (M + 15) >> 4;
    if (wave >= nquads * mtiles) return;        // wave-uniform: EXEC stays full
    const int mt = wave / nquads;
    const int qt = wave % nquads;

    const int arow = (mt << 4) + (lane & 15);
    int col[QT], wrow[QT];
#pragma unroll
    for (int j = 0; j < QT; ++j) {
        col[j]  = ((qt * QT + j) << 4) + (lane & 15);
        wrow[j] = (col[j] < N) ? col[j] : 0;
    }

    v8f acc[QT];
#pragma unroll
    for (int j = 0; j < QT; ++j) acc[j] = (v8f){};

    wmma_quad_k(acc, A0, lda0, K0, W0, (size_t)ldw0, arow, wrow, lane);
    if (W1 && K1 > 0)
        wmma_quad_k(acc, A1, lda1, K1, W1, (size_t)ldw1, arow, wrow, lane);

    const int mbase = (mt << 4) + ((lane < 16) ? 0 : 8);  // C layout 7.12.2
#pragma unroll
    for (int j = 0; j < QT; ++j) {
        if (col[j] >= N) continue;
        const float bv = bias ? bias[col[j]] : 0.0f;
#pragma unroll
        for (int r = 0; r < 8; ++r) {
            float v = acc[j][r] + bv;
            if (mode == 1) v = (v > 0.0f) ? v : 0.01f * v;
            const size_t idx = (size_t)(mbase + r) * ldc + col[j];
            if (C)   C[idx]   = v;
            if (Cbf) Cbf[idx] = (bf16_t)v;
        }
    }
}

// ---------------- small helper kernels ----------------
__global__ void f32_to_bf16_k(const float* __restrict__ s, bf16_t* __restrict__ d, int n) {
    int i = blockIdx.x * blockDim.x + threadIdx.x;
    if (i < n) d[i] = (bf16_t)s[i];
}
__global__ void add_vec_k(const float* a, const float* b, float* o, int n) {
    int i = blockIdx.x * blockDim.x + threadIdx.x;
    if (i < n) o[i] = a[i] + b[i];
}
__global__ void zero_f32_k(float* p, int n) {
    int i = blockIdx.x * blockDim.x + threadIdx.x;
    if (i < n) p[i] = 0.0f;
}
__global__ void zero_bf16_k(bf16_t* p, int n) {
    int i = blockIdx.x * blockDim.x + threadIdx.x;
    if (i < n) p[i] = (bf16_t)0.0f;
}
// x_bf = [emb_bf[tok] | ctx_bf]
__global__ void embed_concat_k(const int* __restrict__ tok,
                               const bf16_t* __restrict__ emb_bf,
                               const bf16_t* __restrict__ ctx_bf,
                               bf16_t* __restrict__ x_bf) {
    int i = blockIdx.x * blockDim.x + threadIdx.x;      // BB * (DD+VV)
    if (i >= BB * (DD + VV)) return;
    int b = i / (DD + VV), j = i % (DD + VV);
    x_bf[(size_t)b * (DD + VV) + j] =
        (j < DD) ? emb_bf[(size_t)tok[b] * DD + j] : ctx_bf[b * VV + (j - DD)];
}
__device__ __forceinline__ float sigmoidf_(float x) { return 1.0f / (1.0f + __expf(-x)); }

__global__ void lstm_pw_k(const float* __restrict__ gates,
                          const float* __restrict__ c_in,
                          bf16_t* __restrict__ h_bf, float* __restrict__ c_out) {
    int i = blockIdx.x * blockDim.x + threadIdx.x;      // BB * DD
    if (i >= BB * DD) return;
    int b = i >> 9, d = i & (DD - 1);
    const float* g = gates + (size_t)b * 4 * DD;
    float ig = sigmoidf_(g[d]);
    float fg = sigmoidf_(g[DD + d]);
    float gg = tanhf(g[2 * DD + d]);
    float og = sigmoidf_(g[3 * DD + d]);
    float c  = fg * c_in[i] + ig * gg;
    c_out[i] = c;
    h_bf[i]  = (bf16_t)(og * tanhf(c));
}

// masked softmax attention over TENC, one block (256 thr) per batch row
__global__ void attend_k(const float* __restrict__ keys,     // (TENC,B,K)
                         const float* __restrict__ values,   // (TENC,B,V)
                         const int*   __restrict__ utt_len,
                         const float* __restrict__ query,    // (B,K) f32
                         bf16_t* __restrict__ ctx_bf,        // (B,V) bf16
                         float* __restrict__ attn_out)       // (B,TENC)
{
    __shared__ float sh_q[KK];
    __shared__ float sh_e[TENC];
    __shared__ float sh_r[256];
    const int b = blockIdx.x, tid = threadIdx.x;
    for (int i = tid; i < KK; i += 256) sh_q[i] = query[b * KK + i];
    __syncthreads();
    const int len = utt_len[b];
    for (int t = tid; t < TENC; t += 256) {
        float e = -10000.0f;
        if (t < len) {
            const float* kp = keys + ((size_t)t * BB + b) * KK;
            float s = 0.0f;
            for (int k = 0; k < KK; ++k) s += kp[k] * sh_q[k];
            e = s;
        }
        sh_e[t] = e;
    }
    __syncthreads();
    float m = fmaxf(sh_e[tid], sh_e[tid + 256]);
    sh_r[tid] = m; __syncthreads();
    for (int s = 128; s > 0; s >>= 1) { if (tid < s) sh_r[tid] = fmaxf(sh_r[tid], sh_r[tid + s]); __syncthreads(); }
    const float emax = sh_r[0]; __syncthreads();
    float lsum = 0.0f;
    for (int t = tid; t < TENC; t += 256) {
        float ex = (t < len) ? __expf(sh_e[t] - emax) : 0.0f;
        sh_e[t] = ex; lsum += ex;
    }
    sh_r[tid] = lsum; __syncthreads();
    for (int s = 128; s > 0; s >>= 1) { if (tid < s) sh_r[tid] += sh_r[tid + s]; __syncthreads(); }
    const float inv = 1.0f / sh_r[0]; __syncthreads();
    for (int t = tid; t < TENC; t += 256) { sh_e[t] *= inv; attn_out[(size_t)b * TENC + t] = sh_e[t]; }
    __syncthreads();
    float acc = 0.0f;
    const int v = tid & (VV - 1), t0 = tid >> 7;
    for (int t = t0; t < TENC; t += 2)
        acc += sh_e[t] * values[((size_t)t * BB + b) * VV + v];
    sh_r[tid] = acc; __syncthreads();
    if (tid < VV) ctx_bf[b * VV + tid] = (bf16_t)(sh_r[tid] + sh_r[tid + VV]);
}

__device__ __forceinline__ float hash_uniform(unsigned t, unsigned b, unsigned v) {
    unsigned long long x = ((unsigned long long)t << 40) ^ ((unsigned long long)b << 20)
                         ^ (unsigned long long)v ^ 0x9E3779B97F4A7C15ull;
    x ^= x >> 30; x *= 0xBF58476D1CE4E5B9ull;
    x ^= x >> 27; x *= 0x94D049BB133111EBull;
    x ^= x >> 31;
    return (float)((x >> 40) * (1.0 / 16777216.0));
}

__global__ void gumbel_argmax_k(const float* __restrict__ logits, // (B,VOCAB)
                                int step, float* __restrict__ gen) {
    const int b = blockIdx.x, tid = threadIdx.x;
    const float* lr = logits + (size_t)b * VOCAB;
    float best = -1e30f; int bi = 0;
    for (int v = tid; v < VOCAB; v += 256) {
        float u = hash_uniform((unsigned)step, (unsigned)b, (unsigned)v);
        float g = -logf(1e-10f - logf(u + 1e-10f));
        float val = lr[v] + g;
        if (val > best) { best = val; bi = v; }
    }
    __shared__ float sv[256]; __shared__ int si[256];
    sv[tid] = best; si[tid] = bi; __syncthreads();
    for (int s = 128; s > 0; s >>= 1) {
        if (tid < s) {
            if (sv[tid + s] > sv[tid] || (sv[tid + s] == sv[tid] && si[tid + s] < si[tid])) {
                sv[tid] = sv[tid + s]; si[tid] = si[tid + s];
            }
        }
        __syncthreads();
    }
    if (tid == 0) gen[b] = (float)si[0];
}

// =====================================================================
// Host side
// =====================================================================
static inline int cdiv(int a, int b) { return (a + b - 1) / b; }

extern "C" void kernel_launch(void* const* d_in, const int* in_sizes, int n_in,
                              void* d_out, int out_size, void* d_ws, size_t ws_size,
                              hipStream_t stream) {
    (void)in_sizes; (void)n_in; (void)out_size; (void)ws_size;
    // ---- inputs (setup_inputs order) ----
    const int*   inputs   = (const int*)  d_in[0];
    const float* keys     = (const float*)d_in[2];
    const float* values   = (const float*)d_in[3];
    const int*   utt_len  = (const int*)  d_in[4];
    const float* emb      = (const float*)d_in[5];
    const float* w_ih[3]  = { (const float*)d_in[6],  (const float*)d_in[10], (const float*)d_in[14] };
    const float* w_hh[3]  = { (const float*)d_in[7],  (const float*)d_in[11], (const float*)d_in[15] };
    const float* b_ih[3]  = { (const float*)d_in[8],  (const float*)d_in[12], (const float*)d_in[16] };
    const float* b_hh[3]  = { (const float*)d_in[9],  (const float*)d_in[13], (const float*)d_in[17] };
    const float* q_w      = (const float*)d_in[18];
    const float* q_b      = (const float*)d_in[19];
    const float* p1_w     = (const float*)d_in[20];
    const float* p1_b     = (const float*)d_in[21];
    const float* p2_b     = (const float*)d_in[22];

    // ---- output layout: logits | attns | generateds ----
    float* out_logits = (float*)d_out;
    float* out_attns  = out_logits + (size_t)TDEC * BB * VOCAB;
    float* out_gen    = out_attns  + (size_t)TDEC * BB * TENC;

    // ---- workspace carve-up ----
    char* p = (char*)d_ws;
    auto carve = [&](size_t bytes) { char* r = p; p += (bytes + 255) & ~(size_t)255; return (void*)r; };
    const int in_dims[3] = { DD + VV, DD, DD };
    bf16_t* wih_bf[3]; bf16_t* whh_bf[3];
    for (int l = 0; l < 3; ++l) {
        wih_bf[l] = (bf16_t*)carve((size_t)4 * DD * in_dims[l] * sizeof(bf16_t));
        whh_bf[l] = (bf16_t*)carve((size_t)4 * DD * DD * sizeof(bf16_t));
    }
    bf16_t* qw_bf  = (bf16_t*)carve((size_t)KK * DD * sizeof(bf16_t));
    bf16_t* p1_bf  = (bf16_t*)carve((size_t)DD * (DD + VV) * sizeof(bf16_t));
    bf16_t* emb_bf = (bf16_t*)carve((size_t)VOCAB * DD * sizeof(bf16_t));
    float* bsum[3];
    for (int l = 0; l < 3; ++l) bsum[l] = (float*)carve((size_t)4 * DD * sizeof(float));

    // f32 state block (c1,c2,c3,query) -- zeroed in one launch
    float* fstates = (float*)carve((size_t)(3 * BB * DD + BB * KK) * sizeof(float));
    float* c1 = fstates;          float* c2 = c1 + BB * DD;
    float* c3 = c2 + BB * DD;     float* query = c3 + BB * DD;
    // bf16 state block (h1,h2,h3,ctx) -- zeroed in one launch
    bf16_t* bstates = (bf16_t*)carve((size_t)(3 * BB * DD + BB * VV) * sizeof(bf16_t));
    bf16_t* h1b = bstates;        bf16_t* h2b = h1b + BB * DD;
    bf16_t* h3b = h2b + BB * DD;  bf16_t* ctxb = h3b + BB * DD;

    bf16_t* x_bf  = (bf16_t*)carve((size_t)BB * (DD + VV) * sizeof(bf16_t));
    float*  gates = (float*) carve((size_t)BB * 4 * DD * sizeof(float));
    bf16_t* hid_bf = (bf16_t*)carve((size_t)BB * DD * sizeof(bf16_t));
    float*  attn_scratch = (float*)carve((size_t)BB * TENC * sizeof(float));

    const int T256 = 256;
    auto cvt = [&](const float* s, bf16_t* d, int n) {
        f32_to_bf16_k<<<cdiv(n, T256), T256, 0, stream>>>(s, d, n);
    };
    // ---- one-time (per launch) weight conversion ----
    for (int l = 0; l < 3; ++l) {
        cvt(w_ih[l], wih_bf[l], 4 * DD * in_dims[l]);
        cvt(w_hh[l], whh_bf[l], 4 * DD * DD);
        add_vec_k<<<cdiv(4 * DD, T256), T256, 0, stream>>>(b_ih[l], b_hh[l], bsum[l], 4 * DD);
    }
    cvt(q_w, qw_bf, KK * DD);
    cvt(p1_w, p1_bf, DD * (DD + VV));
    cvt(emb, emb_bf, VOCAB * DD);

    // ---- zero initial state ----
    {
        int nf = 3 * BB * DD + BB * KK;
        zero_f32_k<<<cdiv(nf, T256), T256, 0, stream>>>(fstates, nf);
        int nb = 3 * BB * DD + BB * VV;
        zero_bf16_k<<<cdiv(nb, T256), T256, 0, stream>>>(bstates, nb);
    }

    // GEMM launch helper: 8 waves per block, each wave = 16 rows x 4 tiles
    auto gemm = [&](const bf16_t* A0, int lda0, int K0, const bf16_t* W0, int ldw0,
                    const bf16_t* A1, int lda1, int K1, const bf16_t* W1, int ldw1,
                    const float* bias, float* C, bf16_t* Cbf, int ldc,
                    int M, int N, int mode) {
        int waves = cdiv(M, 16) * cdiv(cdiv(N, 16), QT);
        wmma_gemm_dual<<<cdiv(waves, 8), 256, 0, stream>>>(
            A0, lda0, K0, W0, ldw0, A1, lda1, K1, W1, ldw1,
            bias, C, Cbf, ldc, M, N, mode);
    };

    // ---- ctx0 = attend(zero query) ----
    attend_k<<<BB, 256, 0, stream>>>(keys, values, utt_len, query, ctxb, attn_scratch);

    // ---- decoder scan ----
    for (int t = 0; t < TDEC; ++t) {
        embed_concat_k<<<cdiv(BB * (DD + VV), T256), T256, 0, stream>>>(
            inputs + (size_t)t * BB, emb_bf, ctxb, x_bf);

        // layer 0: gates = x@w_ih0^T + h1@w_hh0^T + b
        gemm(x_bf, DD + VV, DD + VV, wih_bf[0], DD + VV,
             h1b, DD, DD, whh_bf[0], DD, bsum[0], gates, nullptr, 4 * DD, BB, 4 * DD, 0);
        lstm_pw_k<<<cdiv(BB * DD, T256), T256, 0, stream>>>(gates, c1, h1b, c1);
        // layer 1
        gemm(h1b, DD, DD, wih_bf[1], DD,
             h2b, DD, DD, whh_bf[1], DD, bsum[1], gates, nullptr, 4 * DD, BB, 4 * DD, 0);
        lstm_pw_k<<<cdiv(BB * DD, T256), T256, 0, stream>>>(gates, c2, h2b, c2);
        // layer 2
        gemm(h2b, DD, DD, wih_bf[2], DD,
             h3b, DD, DD, whh_bf[2], DD, bsum[2], gates, nullptr, 4 * DD, BB, 4 * DD, 0);
        lstm_pw_k<<<cdiv(BB * DD, T256), T256, 0, stream>>>(gates, c3, h3b, c3);

        // query = h3 @ q_w^T + q_b   (f32 out for softmax path)
        gemm(h3b, DD, DD, qw_bf, DD, nullptr, 0, 0, nullptr, 0,
             q_b, query, nullptr, KK, BB, KK, 0);

        // attention -> attns[t], new ctx (bf16)
        attend_k<<<BB, 256, 0, stream>>>(keys, values, utt_len, query, ctxb,
                                         out_attns + (size_t)t * BB * TENC);

        // hid = leaky_relu([h3|ctx] @ p1_w^T + p1_b)  -> bf16 for logits GEMM
        gemm(h3b, DD, DD, p1_bf, DD + VV,
             ctxb, VV, VV, p1_bf + DD, DD + VV, p1_b, nullptr, hid_bf, DD, BB, DD, 1);

        // logits = hid @ emb^T + p2_b  (weight-tied), straight into d_out
        float* lg = out_logits + (size_t)t * BB * VOCAB;
        gemm(hid_bf, DD, DD, emb_bf, DD, nullptr, 0, 0, nullptr, 0,
             p2_b, lg, nullptr, VOCAB, BB, VOCAB, 0);

        // gumbel-noise argmax -> generateds[t]
        gumbel_argmax_k<<<BB, 256, 0, stream>>>(lg, t, out_gen + (size_t)t * BB);
    }
}
